// AudioCondTransformerEncoderLayer_40458591928571
// MI455X (gfx1250) — matrix operand order repaired and checked by reading, not compiled, not run on hardware
//
#include <hip/hip_runtime.h>

// ---------------------------------------------------------------------------
// AudioCondTransformerEncoderLayer for MI455X (gfx1250, wave32, WMMA).
// GEMMs in f16 with f32 accumulation via v_wmma_f32_16x16x32_f16.
// Tile staging uses CDNA5 async global->LDS copies (ASYNCcnt) with
// double-buffered software pipelining.
// ---------------------------------------------------------------------------

typedef _Float16 v16h __attribute__((ext_vector_type(16)));
typedef float    v8f  __attribute__((ext_vector_type(8)));

#define D_MODEL 1024
#define NHEAD   16
#define HEADD   64
#define DFF     4096
#define TMOT    512
#define TAUD    512
#define BATCH   32
#define NTOK    (TMOT * BATCH)   // 16384 token rows
#define LDQKV   3072

static __device__ __forceinline__ float gelu_exact(float x) {
    return 0.5f * x * (1.0f + erff(x * 0.70710678118654752f));
}

static __device__ __forceinline__ void wait_async0() {
#if __has_builtin(__builtin_amdgcn_s_wait_asynccnt)
    __builtin_amdgcn_s_wait_asynccnt(0);
#else
    asm volatile("s_wait_asynccnt 0x0" ::: "memory");
#endif
}

// async copy of 16 bytes: global (64-bit vaddr) -> LDS (32-bit byte offset)
static __device__ __forceinline__ void async_b128(unsigned lds_off, const void* gptr) {
    asm volatile("global_load_async_to_lds_b128 %0, %1, off"
                 :: "v"(lds_off), "v"(gptr) : "memory");
}
// 32-byte variant (two b128s; offset: applies to both LDS and global sides)
static __device__ __forceinline__ void async_b256(unsigned lds_off, const void* gptr) {
    asm volatile("global_load_async_to_lds_b128 %0, %1, off\n\t"
                 "global_load_async_to_lds_b128 %0, %1, off offset:16"
                 :: "v"(lds_off), "v"(gptr) : "memory");
}

// ---------------- cast f32 -> f16 ----------------
__global__ __launch_bounds__(256)
void cast_f32_f16(const float* __restrict__ in, _Float16* __restrict__ out, int n) {
    int i = blockIdx.x * 256 + threadIdx.x;
    if (i < n) out[i] = (_Float16)in[i];
}

// ---------------- temporal + beat bias (TMOT x TAUD) ----------------
__global__ __launch_bounds__(256)
void build_bias(const int* __restrict__ beats, float* __restrict__ bias) {
    int i  = blockIdx.x * 256 + threadIdx.x;      // qt*TAUD + kt
    int kt = i & (TAUD - 1);
    int qt = i >> 9;
    // scale = (TAUD-1)/(TMOT-1) == 1.0 here
    float diff = (float)qt - (float)kt;
    float v = -(diff * diff) * (1.0f / 32.0f);    // 2*sigma^2 = 32
    float bb = 0.0f;
#pragma unroll 1
    for (int n = 0; n < 64; n++) {
        int bf = beats[n];
        if (kt == bf)                      bb = fmaxf(bb, 2.0f);   // BW
        if (bf > 0 && kt == bf - 1)        bb = fmaxf(bb, 1.0f);   // BW/2
        if (bf < TAUD - 1 && kt == bf + 1) bb = fmaxf(bb, 1.0f);
    }
    bias[i] = v + bb;
}

// ---------------- tiled WMMA GEMM: C = A(MxK) * W(NxK)^T + bias ----------------
// Double-buffered LDS tiles filled by async global->LDS copies; copy of tile
// k+1 overlaps the WMMA work on tile k.
// EPI 0: f16 out + bias ; EPI 1: f16 out + bias + exact GELU ; EPI 2: f32 out + bias
template<int EPI>
__global__ __launch_bounds__(256)
void gemm_wmma(const _Float16* __restrict__ A, const _Float16* __restrict__ W,
               const float* __restrict__ bias, void* __restrict__ outp,
               int M, int N, int K, int ldout) {
    __shared__ _Float16 sA[2][128][40];   // 128x32 tile, row pad 40 halfs for banks
    __shared__ _Float16 sB[2][128][40];
    const int tid  = threadIdx.x;
    const int lane = tid & 31, wave = tid >> 5;
    const int wm = wave & 3, wn = wave >> 2;           // 4(M) x 2(N) wave grid
    const int bm = blockIdx.y * 128, bn = blockIdx.x * 128;
    const int r  = lane & 15, kh = lane >> 4;
    const int lrow = tid >> 1;                          // 0..127
    const int lcol = (tid & 1) * 16;                    // 0 or 16 halfs
    v8f c[2][4] = {};

    const _Float16* gA = A + (size_t)(bm + lrow) * K + lcol;
    const _Float16* gB = W + (size_t)(bn + lrow) * K + lcol;
    const unsigned laBase[2] = { (unsigned)(size_t)&sA[0][lrow][lcol],
                                 (unsigned)(size_t)&sA[1][lrow][lcol] };
    const unsigned lbBase[2] = { (unsigned)(size_t)&sB[0][lrow][lcol],
                                 (unsigned)(size_t)&sB[1][lrow][lcol] };
    auto issue_tile = [&](int buf, int k0) {
        async_b256(laBase[buf], gA + k0);
        async_b256(lbBase[buf], gB + k0);
    };

    const int nk = K / 32;
    issue_tile(0, 0);
    for (int kt = 0; kt < nk; kt++) {
        const int buf = kt & 1;
        wait_async0();                    // tile kt resident in LDS (this wave's slice)
        __syncthreads();                  // all slices resident; prev buf free
        if (kt + 1 < nk) issue_tile(buf ^ 1, (kt + 1) * 32);

        v16h a[2], b[4];
#pragma unroll
        for (int i = 0; i < 2; i++) {                   // A frag: contiguous 16-K per half
            const _Float16* p = &sA[buf][wm * 32 + i * 16 + r][kh * 16];
#pragma unroll
            for (int u = 0; u < 8; u++) { a[i][u] = p[u]; a[i][u + 8] = p[u + 8]; }
        }
#pragma unroll
        for (int j = 0; j < 4; j++) {                   // B frag: contiguous 16-K per half
            const _Float16* p = &sB[buf][wn * 64 + j * 16 + r][kh * 16];
#pragma unroll
            for (int u = 0; u < 16; u++) b[j][u] = p[u];
        }
#pragma unroll
        for (int i = 0; i < 2; i++)
#pragma unroll
            for (int j = 0; j < 4; j++)
                c[i][j] = __builtin_amdgcn_wmma_f32_16x16x32_f16(
                    false, a[i], false, b[j], (short)0, c[i][j], false, false);
    }

#pragma unroll
    for (int j = 0; j < 4; j++) {
        const int col = bn + wn * 64 + j * 16 + r;      // C layout: N = lane&15
        const float bv = bias ? bias[col] : 0.0f;
#pragma unroll
        for (int i = 0; i < 2; i++) {
#pragma unroll
            for (int e = 0; e < 8; e++) {
                const int row = bm + wm * 32 + i * 16 + e + 8 * kh;
                float v = c[i][j][e] + bv;
                if (EPI == 1) v = gelu_exact(v);
                if (EPI == 2) ((float*)outp)[(size_t)row * ldout + col] = v;
                else          ((_Float16*)outp)[(size_t)row * ldout + col] = (_Float16)v;
            }
        }
    }
}

// ---------------- V transpose: QKV[t*B+b][2048 + h*64 + d] -> Vt[bh][d][t] ----------------
__global__ __launch_bounds__(256)
void transpose_v(const _Float16* __restrict__ QKV, _Float16* __restrict__ Vt) {
    int i  = blockIdx.x * 256 + threadIdx.x;   // bh*32768 + d*512 + t
    int t  = i & 511;
    int d  = (i >> 9) & 63;
    int bh = i >> 15;
    int b = bh >> 4, h = bh & 15;
    Vt[i] = QKV[(size_t)(t * BATCH + b) * LDQKV + 2 * D_MODEL + h * HEADD + d];
}

// ---------------- flash attention: one wave = one (b,h) x 16 Q-rows ----------------
template<bool HAS_BIAS>
__global__ __launch_bounds__(256)
void attn_kernel(const _Float16* __restrict__ QKV, const _Float16* __restrict__ Vt,
                 _Float16* __restrict__ O, const float* __restrict__ bias) {
    __shared__ float sBias[HAS_BIAS ? 16 * 516 : 16];  // pad 516 to break bank conflicts
    __shared__ _Float16 sP[8][16][40];                  // per-wave P tile (16x32), pad 40
    const int tid = threadIdx.x, lane = tid & 31, wave = tid >> 5;
    const int qb = blockIdx.x;                          // q tile (16 rows), shared by block
    const int bh = blockIdx.y * 8 + wave;               // each wave its own (b,h)
    const int b = bh >> 4, h = bh & 15;
    const int r = lane & 15, kh = lane >> 4;

    if (HAS_BIAS) {
        // async-copy the 16x512 f32 bias tile into (padded) LDS
        for (int cIdx = tid; cIdx < 16 * (TAUD / 4); cIdx += 256) {  // float4 chunks
            const int rr = cIdx >> 7;             // 0..15
            const int cc = (cIdx & 127) << 2;     // 0..508 step 4
            const float* gp = bias + (size_t)(qb * 16 + rr) * TAUD + cc;
            async_b128((unsigned)(size_t)&sBias[rr * 516 + cc], gp);
        }
        wait_async0();
        __syncthreads();
    }

    v16h q0, q1;                                        // Q frags (A-operand, hd 0..31 / 32..63)
    {
        const int t = qb * 16 + r;
        const _Float16* p = QKV + (size_t)(t * BATCH + b) * LDQKV + h * HEADD;
#pragma unroll
        for (int u = 0; u < 8; u++) {
            q0[u] = p[kh * 16 + u];      q0[u + 8] = p[kh * 16 + 8 + u];
            q1[u] = p[32 + kh * 16 + u]; q1[u + 8] = p[40 + kh * 16 + u];
        }
    }
    float m[8], l[8];
#pragma unroll
    for (int e = 0; e < 8; e++) { m[e] = -1e30f; l[e] = 0.0f; }
    v8f o[4] = {};

    for (int kb = 0; kb < TAUD / 32; kb++) {
        float sv[2][8];
#pragma unroll
        for (int st = 0; st < 2; st++) {                // two 16-wide score tiles
            const int kt = kb * 32 + st * 16 + r;       // B-operand: lane = K-row (column of S)
            const _Float16* p = QKV + (size_t)(kt * BATCH + b) * LDQKV + D_MODEL + h * HEADD + kh * 16;
            v16h kf0, kf1;
#pragma unroll
            for (int u = 0; u < 16; u++) { kf0[u] = p[u]; kf1[u] = p[32 + u]; }
            v8f acc = {};
            acc = __builtin_amdgcn_wmma_f32_16x16x32_f16(false, q0, false, kf0, (short)0, acc, false, false);
            acc = __builtin_amdgcn_wmma_f32_16x16x32_f16(false, q1, false, kf1, (short)0, acc, false, false);
#pragma unroll
            for (int e = 0; e < 8; e++) {
                float v = acc[e] * 0.125f;              // 1/sqrt(64)
                if (HAS_BIAS) v += sBias[(e + 8 * kh) * 516 + kb * 32 + st * 16 + r];
                sv[st][e] = v;
            }
        }
        // online softmax; lanes 0-15 own rows 0-7, lanes 16-31 own rows 8-15
#pragma unroll
        for (int e = 0; e < 8; e++) {
            float cur = fmaxf(sv[0][e], sv[1][e]);
#pragma unroll
            for (int off = 1; off < 16; off <<= 1) cur = fmaxf(cur, __shfl_xor(cur, off, 32));
            const float mn   = fmaxf(m[e], cur);
            const float corr = __expf(m[e] - mn);
            const float a0   = __expf(sv[0][e] - mn);
            const float a1   = __expf(sv[1][e] - mn);
            float ps = a0 + a1;
#pragma unroll
            for (int off = 1; off < 16; off <<= 1) ps += __shfl_xor(ps, off, 32);
            l[e] = l[e] * corr + ps;
            m[e] = mn;
#pragma unroll
            for (int j = 0; j < 4; j++) o[j][e] *= corr;
            sP[wave][e + 8 * kh][r]      = (_Float16)a0;   // C-layout -> LDS
            sP[wave][e + 8 * kh][16 + r] = (_Float16)a1;
        }
        __builtin_amdgcn_wave_barrier();
        v16h pf;                                         // reload P as A-operand 16x32
#pragma unroll
        for (int u = 0; u < 8; u++) {
            pf[u]     = sP[wave][r][kh * 16 + u];
            pf[u + 8] = sP[wave][r][kh * 16 + 8 + u];
        }
#pragma unroll
        for (int j = 0; j < 4; j++) {                    // V as B-operand from Vt (contiguous)
            const _Float16* p = Vt + ((size_t)bh * HEADD + j * 16 + r) * TAUD + kb * 32 + kh * 16;
            v16h vf;
#pragma unroll
            for (int u = 0; u < 16; u++) vf[u] = p[u];
            o[j] = __builtin_amdgcn_wmma_f32_16x16x32_f16(false, pf, false, vf, (short)0, o[j], false, false);
        }
    }

#pragma unroll
    for (int e = 0; e < 8; e++) {
        const float inv = 1.0f / l[e];
        const int t = qb * 16 + e + 8 * kh;
#pragma unroll
        for (int j = 0; j < 4; j++)
            O[(size_t)(t * BATCH + b) * D_MODEL + h * HEADD + j * 16 + r] = (_Float16)(o[j][e] * inv);
    }
}

// ---------------- fused residual (+tanh gate) + LayerNorm ----------------
__global__ __launch_bounds__(256)
void add_ln(const float* __restrict__ res, const float* __restrict__ y,
            const float* __restrict__ gatep, const float* __restrict__ g,
            const float* __restrict__ bb, float* __restrict__ out32,
            _Float16* __restrict__ out16) {
    __shared__ float red[16];
    const int row = blockIdx.x;
    const int tid = threadIdx.x;
    const float alpha = gatep ? tanhf(gatep[0]) : 1.0f;
    const float* rr = res + (size_t)row * D_MODEL;
    const float* yy = y   + (size_t)row * D_MODEL;
    float v[4];
    float s = 0.0f, sq = 0.0f;
#pragma unroll
    for (int k = 0; k < 4; k++) {
        const float t = rr[tid + k * 256] + alpha * yy[tid + k * 256];
        v[k] = t; s += t; sq += t * t;
    }
#pragma unroll
    for (int off = 1; off < 32; off <<= 1) { s += __shfl_xor(s, off, 32); sq += __shfl_xor(sq, off, 32); }
    if ((tid & 31) == 0) { red[tid >> 5] = s; red[8 + (tid >> 5)] = sq; }
    __syncthreads();
    s = 0.0f; sq = 0.0f;
#pragma unroll
    for (int w = 0; w < 8; w++) { s += red[w]; sq += red[8 + w]; }
    const float mean = s * (1.0f / 1024.0f);
    const float var  = sq * (1.0f / 1024.0f) - mean * mean;
    const float inv  = rsqrtf(var + 1e-5f);
#pragma unroll
    for (int k = 0; k < 4; k++) {
        const int cidx = tid + k * 256;
        const float ov = (v[k] - mean) * inv * g[cidx] + bb[cidx];
        out32[(size_t)row * D_MODEL + cidx] = ov;
        if (out16) out16[(size_t)row * D_MODEL + cidx] = (_Float16)ov;
    }
}

// ---------------------------------------------------------------------------
extern "C" void kernel_launch(void* const* d_in, const int* in_sizes, int n_in,
                              void* d_out, int out_size, void* d_ws, size_t ws_size,
                              hipStream_t stream) {
    const float* src      = (const float*)d_in[0];
    const float* audio    = (const float*)d_in[1];
    const int*   beats    = (const int*)d_in[2];
    const float* sa_in_w  = (const float*)d_in[3];
    const float* sa_in_b  = (const float*)d_in[4];
    const float* sa_out_w = (const float*)d_in[5];
    const float* sa_out_b = (const float*)d_in[6];
    const float* ca_in_w  = (const float*)d_in[7];
    const float* ca_in_b  = (const float*)d_in[8];
    const float* ca_out_w = (const float*)d_in[9];
    const float* ca_out_b = (const float*)d_in[10];
    const float* gate     = (const float*)d_in[11];
    const float* n1_g = (const float*)d_in[12]; const float* n1_b = (const float*)d_in[13];
    const float* nc_g = (const float*)d_in[14]; const float* nc_b = (const float*)d_in[15];
    const float* n2_g = (const float*)d_in[16]; const float* n2_b = (const float*)d_in[17];
    const float* lin1_w = (const float*)d_in[18]; const float* lin1_b = (const float*)d_in[19];
    const float* lin2_w = (const float*)d_in[20]; const float* lin2_b = (const float*)d_in[21];

    char* ws = (char*)d_ws;
    size_t off = 0;
    auto take = [&](size_t bytes) -> char* {
        char* p = ws + off; off += (bytes + 255) & ~(size_t)255; return p;
    };
    _Float16* wSaIn  = (_Float16*)take((size_t)3 * D_MODEL * D_MODEL * 2);
    _Float16* wSaOut = (_Float16*)take((size_t)D_MODEL * D_MODEL * 2);
    _Float16* wCaIn  = (_Float16*)take((size_t)3 * D_MODEL * D_MODEL * 2);
    _Float16* wCaOut = (_Float16*)take((size_t)D_MODEL * D_MODEL * 2);
    _Float16* wL1    = (_Float16*)take((size_t)DFF * D_MODEL * 2);
    _Float16* wL2    = (_Float16*)take((size_t)D_MODEL * DFF * 2);
    float*    biasQK = (float*)take((size_t)TMOT * TAUD * 4);
    _Float16* aH     = (_Float16*)take((size_t)NTOK * D_MODEL * 2);   // src_h -> x_h -> ...
    _Float16* audH   = (_Float16*)take((size_t)NTOK * D_MODEL * 2);
    char*     qkvReg = take((size_t)NTOK * LDQKV * 2);                // 100,663,296 B
    _Float16* qkv    = (_Float16*)qkvReg;
    _Float16* vt     = (_Float16*)take((size_t)BATCH * NHEAD * HEADD * TAUD * 2); // 33,554,432 B
    _Float16* hff    = (_Float16*)qkvReg;  // alias: QKV+Vt (contiguous 134,217,728 B) reused for FFN hidden
    _Float16* o16    = (_Float16*)take((size_t)NTOK * D_MODEL * 2);
    float*    t32    = (float*)take((size_t)NTOK * D_MODEL * 4);
    float*    x32    = (float*)take((size_t)NTOK * D_MODEL * 4);
    (void)ws_size; (void)n_in; (void)in_sizes; (void)out_size;

    auto castN = [&](const float* in, _Float16* out, int n) {
        cast_f32_f16<<<(n + 255) / 256, 256, 0, stream>>>(in, out, n);
    };
    castN(sa_in_w,  wSaIn,  3 * D_MODEL * D_MODEL);
    castN(sa_out_w, wSaOut, D_MODEL * D_MODEL);
    castN(ca_in_w,  wCaIn,  3 * D_MODEL * D_MODEL);
    castN(ca_out_w, wCaOut, D_MODEL * D_MODEL);
    castN(lin1_w,   wL1,    DFF * D_MODEL);
    castN(lin2_w,   wL2,    D_MODEL * DFF);
    castN(src,      aH,     NTOK * D_MODEL);
    castN(audio,    audH,   NTOK * D_MODEL);
    build_bias<<<(TMOT * TAUD) / 256, 256, 0, stream>>>(beats, biasQK);

    // ---- self attention ----
    gemm_wmma<0><<<dim3(3072 / 128, NTOK / 128), 256, 0, stream>>>(
        aH, wSaIn, sa_in_b, qkv, NTOK, 3072, D_MODEL, LDQKV);
    transpose_v<<<(BATCH * NHEAD * HEADD * TAUD) / 256, 256, 0, stream>>>(qkv, vt);
    attn_kernel<false><<<dim3(TMOT / 16, (BATCH * NHEAD) / 8), 256, 0, stream>>>(qkv, vt, o16, nullptr);
    gemm_wmma<2><<<dim3(D_MODEL / 128, NTOK / 128), 256, 0, stream>>>(
        o16, wSaOut, sa_out_b, t32, NTOK, D_MODEL, D_MODEL, D_MODEL);
    add_ln<<<NTOK, 256, 0, stream>>>(src, t32, nullptr, n1_g, n1_b, x32, aH);

    // ---- biased cross attention (Q from x, K/V from audio) ----
    gemm_wmma<0><<<dim3(D_MODEL / 128, NTOK / 128), 256, 0, stream>>>(
        aH, wCaIn, ca_in_b, qkv, NTOK, D_MODEL, D_MODEL, LDQKV);
    gemm_wmma<0><<<dim3((2 * D_MODEL) / 128, NTOK / 128), 256, 0, stream>>>(
        audH, wCaIn + (size_t)D_MODEL * D_MODEL, ca_in_b + D_MODEL,
        qkv + D_MODEL, NTOK, 2 * D_MODEL, D_MODEL, LDQKV);
    transpose_v<<<(BATCH * NHEAD * HEADD * TAUD) / 256, 256, 0, stream>>>(qkv, vt);
    attn_kernel<true><<<dim3(TMOT / 16, (BATCH * NHEAD) / 8), 256, 0, stream>>>(qkv, vt, o16, biasQK);
    gemm_wmma<2><<<dim3(D_MODEL / 128, NTOK / 128), 256, 0, stream>>>(
        o16, wCaOut, ca_out_b, t32, NTOK, D_MODEL, D_MODEL, D_MODEL);
    add_ln<<<NTOK, 256, 0, stream>>>(x32, t32, gate, nc_g, nc_b, x32, aH);

    // ---- FFN (lin1 + exact GELU fused, lin2, final LN) ----
    gemm_wmma<1><<<dim3(DFF / 128, NTOK / 128), 256, 0, stream>>>(
        aH, wL1, lin1_b, hff, NTOK, DFF, D_MODEL, DFF);
    gemm_wmma<2><<<dim3(D_MODEL / 128, NTOK / 128), 256, 0, stream>>>(
        hff, wL2, lin2_b, t32, NTOK, D_MODEL, DFF, D_MODEL);
    add_ln<<<NTOK, 256, 0, stream>>>(x32, t32, nullptr, n2_g, n2_b, (float*)d_out, nullptr);
}
